// WSTAL_27882927686266
// MI455X (gfx1250) — matrix operand order repaired and checked by reading, not compiled, not run on hardware
//
#include <hip/hip_runtime.h>
#include <hip/hip_bf16.h>
#include <math.h>

// ---------------------------------------------------------------------------
// WSTAL forward for MI455X (gfx1250, wave32, WMMA).
// Main GEMM uses split-f16 (hi+lo) V_WMMA_F32_16X16X32_F16 for ~fp32 accuracy,
// software-pipelined (register prefetch of next K-tile overlaps WMMA compute).
// random_walk solve replaced by exact Woodbury identity (A is rank-8).
// ---------------------------------------------------------------------------

typedef __attribute__((ext_vector_type(16))) _Float16 v16h;
typedef __attribute__((ext_vector_type(8)))  _Float16 v8h;
typedef __attribute__((ext_vector_type(8)))  float    v8f;

static constexpr int NB   = 16;
static constexpr int TT   = 750;
static constexpr int DD   = 2048;
static constexpr int KK   = 8;
static constexpr int C1   = 21;
static constexpr int MROW = NB * TT;          // 12000

// d_out float offsets (tuple flattened in return order)
static constexpr size_t O_CA   = 0;
static constexpr size_t O_CW   = 336;
static constexpr size_t O_ATT  = 672;
static constexpr size_t O_FRM  = 12672;
static constexpr size_t MO_CA  = 264672;
static constexpr size_t MO_CW  = 265008;
static constexpr size_t MO_ATT = 265344;
static constexpr size_t MO_FRM = 277344;
static constexpr size_t X_OFF  = 529344;
static constexpr size_t MU_OFF = 25105344;
static constexpr size_t MUP_OFF= 25367488;

// workspace float offsets (~105 MB total)
static constexpr size_t WS_Y2X    = 0;
static constexpr size_t WS_INVX   = 24576000;
static constexpr size_t WS_INVR   = 24588000;
static constexpr size_t WS_INVMU  = 24600000;
static constexpr size_t WS_Z      = 24600128;
static constexpr size_t WS_NZ     = 24696128;
static constexpr size_t WS_SINV   = 24792128;
static constexpr size_t WS_P      = 24793152;
static constexpr size_t WS_Q      = 25055296;
static constexpr size_t WS_NAC    = 25317440;   // 21 nac rows + nfg at row 21
static constexpr size_t WS_CWATT  = 25362496;
static constexpr size_t WS_CASUM  = 25614496;
static constexpr size_t WS_CWSUM  = 25614512;
static constexpr size_t WS_CAFEAT = 25614848;
static constexpr size_t WS_CWFEAT = 25647616;
static constexpr size_t WS_CASCR  = 26335744;
static constexpr size_t WS_CWSCR  = 26336080;

__device__ __forceinline__ float wred(float v) {
#pragma unroll
  for (int s = 16; s; s >>= 1) v += __shfl_xor(v, s, 32);
  return v;
}

// ---------------------------------------------------------------------------
// Kernel 1: xe = relu(x @ W + b), split-f16 WMMA GEMM, register-prefetch
// pipelined. Block tile 128(M) x 64(N), K step 32, 8 waves, 2x2 frags/wave.
// ---------------------------------------------------------------------------
#define LSTR 40   // LDS row stride in halves: 80B, 16B-aligned, conflict-free

__global__ __launch_bounds__(256) void k_emb_gemm(
    const float* __restrict__ X, const float* __restrict__ Wm,
    const float* __restrict__ bias, float* __restrict__ out)
{
  __shared__ __align__(16) _Float16 Ah[128 * LSTR];
  __shared__ __align__(16) _Float16 Al[128 * LSTR];
  __shared__ __align__(16) _Float16 Bh[64 * LSTR];
  __shared__ __align__(16) _Float16 Bl[64 * LSTR];

  const int tid  = threadIdx.x;
  const int lane = tid & 31;
  const int wv   = tid >> 5;
  const int wm   = wv & 3;          // 0..3  -> M offset wm*32
  const int wn   = wv >> 2;         // 0..1  -> N offset wn*32
  const int bm   = blockIdx.x * 128;
  const int bn   = blockIdx.y * 64;

  v8f acc[2][2] = {};

  const int arow = tid >> 1;            // 0..127
  const int acol = (tid & 1) * 16;      // 0 or 16
  const int bk   = tid >> 3;            // 0..31
  const int bc   = (tid & 7) * 8;       // 0..56

  int gm = bm + arow; if (gm > MROW - 1) gm = MROW - 1;
  const float* aptr = X + (size_t)gm * DD + acol;
  const float* bptr = Wm + (size_t)bk * DD + bn + bc;

  float4 ra[4];   // prefetched A: 16 floats
  float4 rb[2];   // prefetched B: 8 floats
  // prologue: fetch K-tile 0
#pragma unroll
  for (int i = 0; i < 4; ++i) ra[i] = *(const float4*)(aptr + 4 * i);
  rb[0] = *(const float4*)(bptr);
  rb[1] = *(const float4*)(bptr + 4);

  const int mhalf = (lane >> 4) * 8;   // A: chunk0 k=mhalf..+7, chunk1 = 16+mhalf..+7
  const int kb    = (lane >> 4) * 8;   // B: chunk0 k=kb..+7, chunk1 = 16+kb..+7

  for (int kk = 0; kk < DD; kk += 32) {
    __syncthreads();   // prior compute finished reading LDS
    { // convert + store prefetched A tile (hi/lo split)
      const float* fp = (const float*)&ra[0];
#pragma unroll
      for (int j = 0; j < 16; ++j) {
        float v = fp[j];
        _Float16 h = (_Float16)v;
        Ah[arow * LSTR + acol + j] = h;
        Al[arow * LSTR + acol + j] = (_Float16)(v - (float)h);
      }
    }
    { // convert + store prefetched B tile transposed [n][k]
      const float* fp = (const float*)&rb[0];
#pragma unroll
      for (int j = 0; j < 8; ++j) {
        float v = fp[j];
        _Float16 h = (_Float16)v;
        Bh[(bc + j) * LSTR + bk] = h;
        Bl[(bc + j) * LSTR + bk] = (_Float16)(v - (float)h);
      }
    }
    __syncthreads();

    // issue next K-tile's global loads early; latency hides behind WMMAs
    if (kk + 32 < DD) {
      const float* ap = aptr + kk + 32;
      const float* bp = bptr + (size_t)(kk + 32) * DD;
#pragma unroll
      for (int i = 0; i < 4; ++i) ra[i] = *(const float4*)(ap + 4 * i);
      rb[0] = *(const float4*)(bp);
      rb[1] = *(const float4*)(bp + 4);
    }

    auto loadA = [&](int mi, const _Float16* base) -> v16h {
      int r = wm * 32 + mi * 16 + (lane & 15);
      union { v16h v; v8h h[2]; } u;
      u.h[0] = *(const v8h*)&base[r * LSTR + mhalf];
      u.h[1] = *(const v8h*)&base[r * LSTR + 16 + mhalf];
      return u.v;
    };
    auto loadB = [&](int ni, const _Float16* base) -> v16h {
      int r = wn * 32 + ni * 16 + (lane & 15);
      union { v16h v; v8h h[2]; } u;
      u.h[0] = *(const v8h*)&base[r * LSTR + kb];
      u.h[1] = *(const v8h*)&base[r * LSTR + kb + 16];
      return u.v;
    };
    v16h aH[2], aL[2], bH[2], bL[2];
    aH[0] = loadA(0, Ah); aH[1] = loadA(1, Ah);
    aL[0] = loadA(0, Al); aL[1] = loadA(1, Al);
    bH[0] = loadB(0, Bh); bH[1] = loadB(1, Bh);
    bL[0] = loadB(0, Bl); bL[1] = loadB(1, Bl);

#pragma unroll
    for (int i = 0; i < 2; ++i)
#pragma unroll
      for (int j = 0; j < 2; ++j) {
        acc[i][j] = __builtin_amdgcn_wmma_f32_16x16x32_f16(
            false, aH[i], false, bH[j], (short)0, acc[i][j], false, false);
        acc[i][j] = __builtin_amdgcn_wmma_f32_16x16x32_f16(
            false, aH[i], false, bL[j], (short)0, acc[i][j], false, false);
        acc[i][j] = __builtin_amdgcn_wmma_f32_16x16x32_f16(
            false, aL[i], false, bH[j], (short)0, acc[i][j], false, false);
      }
  }

  // epilogue: D layout -> m = fragbase + (lane>>4)*8 + r, n = base + (lane&15)
  // MROW % 16 == 0, so a 16x16 fragment is valid all-or-nothing: per-fragment
  // guard keeps the store path branchless.
#pragma unroll
  for (int i = 0; i < 2; ++i)
#pragma unroll
    for (int j = 0; j < 2; ++j) {
      int fragm = bm + wm * 32 + i * 16;
      if (fragm + 16 > MROW) continue;
      int mbase = fragm + (lane >> 4) * 8;
      int nn    = bn + wn * 32 + j * 16 + (lane & 15);
      float bv  = bias[nn];
#pragma unroll
      for (int r = 0; r < 8; ++r) {
        float v = acc[i][j][r] + bv;
        out[(size_t)(mbase + r) * DD + nn] = v > 0.f ? v : 0.f;
      }
    }
}

// ---------------------------------------------------------------------------
// small helper kernels
// ---------------------------------------------------------------------------
__global__ void k_row_invnorm(const float* __restrict__ X, float* __restrict__ inv,
                              int rows) {
  int row = blockIdx.x * 8 + (threadIdx.x >> 5);
  if (row >= rows) return;
  int lane = threadIdx.x & 31;
  const float* p = X + (size_t)row * DD;
  float s = 0.f;
  for (int i = lane * 4; i < DD; i += 128) {
    float4 v = *(const float4*)(p + i);
    s += v.x * v.x + v.y * v.y + v.z * v.z + v.w * v.w;
  }
  s = wred(s);
  if (lane == 0) inv[row] = 1.0f / (sqrtf(s) + 1e-9f);
}

__global__ void k_bcast_mu(const float* __restrict__ mup, float* __restrict__ mu) {
  int idx = blockIdx.x * 256 + threadIdx.x;   // 16*8*2048
  mu[idx] = mup[idx % (KK * DD)];
}

// z[n,k,t] = softmax_k( 5 * invmu[n,k] * invx[n,t] * dot(mu[n,k,:], xe[n,t,:]) )
__global__ __launch_bounds__(256) void k_zlogits(
    const float* __restrict__ mu, const float* __restrict__ xe,
    const float* __restrict__ invmu, const float* __restrict__ invx,
    float* __restrict__ z) {
  int row = blockIdx.x * 8 + (threadIdx.x >> 5);
  if (row >= MROW) return;
  int n = row / TT, t = row % TT;
  int lane = threadIdx.x & 31;
  const float* xr = xe + (size_t)row * DD;
  const float* mb = mu + (size_t)n * KK * DD;
  float acc[KK] = {};
  for (int i = lane * 4; i < DD; i += 128) {
    float4 xv = *(const float4*)(xr + i);
#pragma unroll
    for (int k = 0; k < KK; ++k) {
      float4 mv = *(const float4*)(mb + k * DD + i);
      acc[k] += xv.x * mv.x + xv.y * mv.y + xv.z * mv.z + xv.w * mv.w;
    }
  }
  float ivx = invx[row];
  float L[KK], mx = -1e30f;
#pragma unroll
  for (int k = 0; k < KK; ++k) {
    float s = wred(acc[k]);
    L[k] = 5.0f * s * invmu[n * KK + k] * ivx;
    mx = fmaxf(mx, L[k]);
  }
  float sum = 0.f;
#pragma unroll
  for (int k = 0; k < KK; ++k) { L[k] = expf(L[k] - mx); sum += L[k]; }
  float r = 1.0f / sum;
  if (lane == 0)
    for (int k = 0; k < KK; ++k) z[((size_t)n * KK + k) * TT + t] = L[k] * r;
}

__global__ void k_nz(const float* __restrict__ z, float* __restrict__ nz) {
  int row = blockIdx.x * 8 + (threadIdx.x >> 5);  // n*8+k, 128 rows
  if (row >= NB * KK) return;
  int lane = threadIdx.x & 31;
  const float* p = z + (size_t)row * TT;
  float s = 0.f;
  for (int i = lane; i < TT; i += 32) s += p[i];
  s = wred(s);
  float r = 1.0f / (s + 1e-9f);
  float* q = nz + (size_t)row * TT;
  for (int i = lane; i < TT; i += 32) q[i] = p[i] * r;
}

// out[n,k,d] = sum_t coef[n,k,t] * feat[n,t,d]   (mu-update and P = NZ@B)
__global__ __launch_bounds__(256) void k_atx(const float* __restrict__ coef,
                                             const float* __restrict__ feat,
                                             float* __restrict__ out) {
  __shared__ float c[KK * TT];
  int n = blockIdx.x >> 3, chunk = blockIdx.x & 7;
  for (int i = threadIdx.x; i < KK * TT; i += 256) c[i] = coef[(size_t)n * KK * TT + i];
  __syncthreads();
  int d = chunk * 256 + threadIdx.x;
  const float* f = feat + (size_t)n * TT * DD + d;
  float acc[KK] = {};
  for (int t = 0; t < TT; ++t) {
    float xv = f[(size_t)t * DD];
#pragma unroll
    for (int k = 0; k < KK; ++k) acc[k] += c[k * TT + t] * xv;
  }
#pragma unroll
  for (int k = 0; k < KK; ++k) out[((size_t)n * KK + k) * DD + d] = acc[k];
}

// S = I8 - 0.25 * NZ @ Z^T ; invert (diag-dominant, pivot-free GJ)
__global__ void k_smat(const float* __restrict__ nz, const float* __restrict__ z,
                       float* __restrict__ sinv) {
  __shared__ float S[64], Inv[64];
  int n = blockIdx.x;
  int i = threadIdx.x >> 3, j = threadIdx.x & 7;
  const float* pz = z + (size_t)n * KK * TT;
  const float* pn = nz + (size_t)n * KK * TT;
  float s = 0.f;
  for (int t = 0; t < TT; ++t) s += pn[i * TT + t] * pz[j * TT + t];
  S[i * 8 + j] = (i == j ? 1.0f : 0.0f) - 0.25f * s;
  Inv[i * 8 + j] = (i == j) ? 1.0f : 0.0f;
  __syncthreads();
  if (threadIdx.x == 0) {
    for (int p = 0; p < 8; ++p) {
      float d = 1.0f / S[p * 8 + p];
      for (int c = 0; c < 8; ++c) { S[p * 8 + c] *= d; Inv[p * 8 + c] *= d; }
      for (int r = 0; r < 8; ++r) if (r != p) {
        float f = S[r * 8 + p];
        for (int c = 0; c < 8; ++c) { S[r * 8 + c] -= f * S[p * 8 + c]; Inv[r * 8 + c] -= f * Inv[p * 8 + c]; }
      }
    }
  }
  __syncthreads();
  sinv[n * 64 + threadIdx.x] = Inv[threadIdx.x];
}

// B = 0.5 * Z^T @ mu + xe
__global__ __launch_bounds__(256) void k_y2x(const float* __restrict__ z,
                                             const float* __restrict__ mu,
                                             const float* __restrict__ xe,
                                             float* __restrict__ y2x) {
  __shared__ float zt[KK];
  int n = blockIdx.x / TT, t = blockIdx.x % TT;
  if (threadIdx.x < KK) zt[threadIdx.x] = z[((size_t)n * KK + threadIdx.x) * TT + t];
  __syncthreads();
  const float* mb = mu + (size_t)n * KK * DD;
  size_t rb = ((size_t)n * TT + t) * DD;
  for (int d = threadIdx.x; d < DD; d += 256) {
    float a = 0.f;
#pragma unroll
    for (int k = 0; k < KK; ++k) a += zt[k] * mb[k * DD + d];
    y2x[rb + d] = 0.5f * a + xe[rb + d];
  }
}

// Q = Sinv @ P (per n)
__global__ void k_q(const float* __restrict__ sinv, const float* __restrict__ p,
                    float* __restrict__ q) {
  int n = blockIdx.x >> 3;
  int d = (blockIdx.x & 7) * 256 + threadIdx.x;
  const float* S = sinv + n * 64;
  const float* P = p + (size_t)n * KK * DD;
  float col[KK];
#pragma unroll
  for (int j = 0; j < KK; ++j) col[j] = P[j * DD + d];
#pragma unroll
  for (int i = 0; i < KK; ++i) {
    float a = 0.f;
#pragma unroll
    for (int j = 0; j < KK; ++j) a += S[i * 8 + j] * col[j];
    q[((size_t)n * KK + i) * DD + d] = a;
  }
}

// rx = 0.5*B + 0.125 * Z^T @ Q   (in-place on B)
__global__ __launch_bounds__(256) void k_refine(const float* __restrict__ z,
                                                const float* __restrict__ q,
                                                float* __restrict__ rx) {
  __shared__ float zt[KK];
  int n = blockIdx.x / TT, t = blockIdx.x % TT;
  if (threadIdx.x < KK) zt[threadIdx.x] = z[((size_t)n * KK + threadIdx.x) * TT + t];
  __syncthreads();
  const float* qb = q + (size_t)n * KK * DD;
  size_t rb = ((size_t)n * TT + t) * DD;
  for (int d = threadIdx.x; d < DD; d += 256) {
    float a = 0.f;
#pragma unroll
    for (int i = 0; i < KK; ++i) a += zt[i] * qb[i * DD + d];
    rx[rb + d] = 0.5f * rx[rb + d] + 0.125f * a;
  }
}

// normalized centers: rows 0..20 = l2n(ac), row 21 = l2n(fg)
__global__ void k_centers(const float* __restrict__ ac, const float* __restrict__ fg,
                          float* __restrict__ nacfg) {
  int row = blockIdx.x * 8 + (threadIdx.x >> 5);
  if (row >= 22) return;
  int lane = threadIdx.x & 31;
  const float* src = (row < 21) ? (ac + (size_t)row * DD) : fg;
  float s = 0.f;
  for (int i = lane * 4; i < DD; i += 128) {
    float4 v = *(const float4*)(src + i);
    s += v.x * v.x + v.y * v.y + v.z * v.z + v.w * v.w;
  }
  s = wred(s);
  float r = 1.0f / (sqrtf(s) + 1e-9f);
  float* dst = nacfg + (size_t)row * DD;
  for (int i = lane; i < DD; i += 32) dst[i] = src[i] * r;
}

// frame scores + attentions
__global__ __launch_bounds__(256) void k_frm(const float* __restrict__ feat,
                                             const float* __restrict__ invn,
                                             const float* __restrict__ nacfg,
                                             float* __restrict__ frm,
                                             float* __restrict__ att,
                                             float* __restrict__ cwatt) {
  int row = blockIdx.x * 8 + (threadIdx.x >> 5);
  if (row >= MROW) return;
  int lane = threadIdx.x & 31;
  int n = row / TT, t = row % TT;
  float iv = invn[row];
  const float* xr = feat + (size_t)row * DD;
  float4 xv[16];
#pragma unroll
  for (int i = 0; i < 16; ++i) xv[i] = *(const float4*)(xr + lane * 4 + i * 128);
#pragma unroll 1
  for (int c = 0; c < 22; ++c) {
    const float* cr = nacfg + (size_t)c * DD;
    float s = 0.f;
#pragma unroll
    for (int i = 0; i < 16; ++i) {
      float4 cv = *(const float4*)(cr + lane * 4 + i * 128);
      s += xv[i].x * cv.x + xv[i].y * cv.y + xv[i].z * cv.z + xv[i].w * cv.w;
    }
    s = wred(s) * iv * 20.0f;
    if (lane == 0) {
      if (c < 21) {
        frm[((size_t)n * TT + t) * C1 + c] = s;
        cwatt[((size_t)n * TT + t) * C1 + c] = 1.0f / (1.0f + expf(-s));
      } else {
        att[row] = 1.0f / (1.0f + expf(-s));
      }
    }
  }
}

__global__ void k_att_sums(const float* __restrict__ att, const float* __restrict__ cwatt,
                           float* __restrict__ casum, float* __restrict__ cwsum) {
  __shared__ float red[256];
  int n = blockIdx.x;
  float s = 0.f;
  for (int t = threadIdx.x; t < TT; t += 256) s += att[n * TT + t];
  red[threadIdx.x] = s;
  __syncthreads();
  for (int st = 128; st; st >>= 1) {
    if (threadIdx.x < st) red[threadIdx.x] += red[threadIdx.x + st];
    __syncthreads();
  }
  if (threadIdx.x == 0) casum[n] = red[0];
  if (threadIdx.x < C1) {
    float c = 0.f;
    const float* p = cwatt + (size_t)n * TT * C1 + threadIdx.x;
    for (int t = 0; t < TT; ++t) c += p[t * C1];
    cwsum[n * C1 + threadIdx.x] = c;
  }
}

__global__ __launch_bounds__(256) void k_ca_feat(const float* __restrict__ feat,
                                                 const float* __restrict__ att,
                                                 const float* __restrict__ casum,
                                                 float* __restrict__ cafeat) {
  __shared__ float a[TT];
  int n = blockIdx.x >> 3, chunk = blockIdx.x & 7;
  float r = 1.0f / (casum[n] + 1e-5f);
  for (int i = threadIdx.x; i < TT; i += 256) a[i] = att[n * TT + i] * r;
  __syncthreads();
  int d = chunk * 256 + threadIdx.x;
  const float* f = feat + (size_t)n * TT * DD + d;
  float acc = 0.f;
  for (int t = 0; t < TT; ++t) acc += a[t] * f[(size_t)t * DD];
  cafeat[n * DD + d] = acc;
}

__global__ __launch_bounds__(256) void k_cw_feat(const float* __restrict__ feat,
                                                 const float* __restrict__ cwatt,
                                                 const float* __restrict__ cwsum,
                                                 float* __restrict__ cwfeat) {
  __shared__ float a[125 * C1];
  __shared__ float rinv[C1];
  int n = blockIdx.x >> 3, chunk = blockIdx.x & 7;
  if (threadIdx.x < C1) rinv[threadIdx.x] = 1.0f / (cwsum[n * C1 + threadIdx.x] + 1e-5f);
  __syncthreads();
  int d = chunk * 256 + threadIdx.x;
  const float* f = feat + (size_t)n * TT * DD + d;
  float acc[C1] = {};
  for (int t0 = 0; t0 < TT; t0 += 125) {
    __syncthreads();
    for (int i = threadIdx.x; i < 125 * C1; i += 256)
      a[i] = cwatt[((size_t)n * TT + t0) * C1 + i] * rinv[i % C1];
    __syncthreads();
    for (int tt = 0; tt < 125; ++tt) {
      float xv = f[(size_t)(t0 + tt) * DD];
#pragma unroll
      for (int c = 0; c < C1; ++c) acc[c] += a[tt * C1 + c] * xv;
    }
  }
#pragma unroll
  for (int c = 0; c < C1; ++c) cwfeat[((size_t)n * C1 + c) * DD + d] = acc[c];
}

__global__ __launch_bounds__(256) void k_scores(const float* __restrict__ cafeat,
                                                const float* __restrict__ cwfeat,
                                                const float* __restrict__ nacfg,
                                                float* __restrict__ cascr,
                                                float* __restrict__ cwscr) {
  __shared__ float cadot[24];
  __shared__ float canorm;
  int n = blockIdx.x;
  int wv = threadIdx.x >> 5, lane = threadIdx.x & 31;
  const float* caf = cafeat + n * DD;
  for (int c = wv; c < C1; c += 8) {
    const float* cr = nacfg + (size_t)c * DD;
    const float* cwf = cwfeat + ((size_t)n * C1 + c) * DD;
    float d1 = 0.f, d2 = 0.f, q2 = 0.f;
    for (int i = lane; i < DD; i += 32) {
      float cv = cr[i], wvv = cwf[i];
      d1 += caf[i] * cv; d2 += wvv * cv; q2 += wvv * wvv;
    }
    d1 = wred(d1); d2 = wred(d2); q2 = wred(q2);
    if (lane == 0) {
      cwscr[n * C1 + c] = 20.0f * d2 / (sqrtf(q2) + 1e-9f);
      cadot[c] = d1;
    }
  }
  if (wv == 0) {
    float s = 0.f;
    for (int i = lane; i < DD; i += 32) { float v = caf[i]; s += v * v; }
    s = wred(s);
    if (lane == 0) canorm = sqrtf(s) + 1e-9f;
  }
  __syncthreads();
  if (threadIdx.x < C1) cascr[n * C1 + threadIdx.x] = 20.0f * cadot[threadIdx.x] / canorm;
}

__global__ void k_softmax21(const float* __restrict__ scr, float* __restrict__ out,
                            int rows) {
  int row = blockIdx.x * 8 + (threadIdx.x >> 5);
  if (row >= rows) return;
  int lane = threadIdx.x & 31;
  float v = (lane < C1) ? scr[row * C1 + lane] : -1e30f;
  float m = v;
#pragma unroll
  for (int s = 16; s; s >>= 1) m = fmaxf(m, __shfl_xor(m, s, 32));
  float e = (lane < C1) ? expf(v - m) : 0.0f;
  float sum = wred(e);
  if (lane < C1) out[row * C1 + lane] = e / sum;
}

__global__ void k_mu_pred(const float* __restrict__ mu, const float* __restrict__ nacfg,
                          float* __restrict__ out) {
  int row = blockIdx.x * 8 + (threadIdx.x >> 5);  // 0..127
  if (row >= NB * KK) return;
  int lane = threadIdx.x & 31;
  const float* mr = mu + (size_t)row * DD;
  float s = 0.f;
  for (int i = lane; i < DD; i += 32) { float v = mr[i]; s += v * v; }
  s = wred(s);
  float inv = 1.0f / (sqrtf(s) + 1e-9f);
  float scr[C1], mx = -1e30f;
#pragma unroll 1
  for (int c = 0; c < C1; ++c) {
    const float* cr = nacfg + (size_t)c * DD;
    float d = 0.f;
    for (int i = lane; i < DD; i += 32) d += mr[i] * cr[i];
    d = wred(d) * inv * 20.0f;
    scr[c] = d; mx = fmaxf(mx, d);
  }
  float sum = 0.f;
#pragma unroll
  for (int c = 0; c < C1; ++c) { scr[c] = expf(scr[c] - mx); sum += scr[c]; }
  float r = 1.0f / sum;
  if (lane == 0)
    for (int c = 0; c < C1; ++c) out[row * C1 + c] = scr[c] * r;
}

// ---------------------------------------------------------------------------
extern "C" void kernel_launch(void* const* d_in, const int* in_sizes, int n_in,
                              void* d_out, int out_size, void* d_ws, size_t ws_size,
                              hipStream_t stream) {
  (void)in_sizes; (void)n_in; (void)out_size; (void)ws_size;
  const float* x   = (const float*)d_in[0];
  const float* Wm  = (const float*)d_in[1];
  const float* be  = (const float*)d_in[2];
  const float* mup = (const float*)d_in[3];
  const float* ac  = (const float*)d_in[4];
  const float* fg  = (const float*)d_in[5];
  float* out = (float*)d_out;
  float* ws  = (float*)d_ws;

  float* xe    = out + X_OFF;
  float* mu    = out + MU_OFF;
  float* y2x   = ws + WS_Y2X;
  float* invx  = ws + WS_INVX;
  float* invr  = ws + WS_INVR;
  float* invmu = ws + WS_INVMU;
  float* z     = ws + WS_Z;
  float* nz    = ws + WS_NZ;
  float* sinv  = ws + WS_SINV;
  float* P     = ws + WS_P;
  float* Q     = ws + WS_Q;
  float* nacfg = ws + WS_NAC;
  float* cwatt = ws + WS_CWATT;
  float* casum = ws + WS_CASUM;
  float* cwsum = ws + WS_CWSUM;
  float* cafeat= ws + WS_CAFEAT;
  float* cwfeat= ws + WS_CWFEAT;
  float* cascr = ws + WS_CASCR;
  float* cwscr = ws + WS_CWSCR;

  // 1. embedding GEMM (WMMA): xe = relu(x @ W + b)
  dim3 g1((MROW + 127) / 128, DD / 64, 1);
  k_emb_gemm<<<g1, 256, 0, stream>>>(x, Wm, be, xe);
  k_row_invnorm<<<MROW / 8, 256, 0, stream>>>(xe, invx, MROW);

  // 2. EM (5 iterations)
  k_bcast_mu<<<NB * KK * DD / 256, 256, 0, stream>>>(mup, mu);
  for (int it = 0; it < 5; ++it) {
    k_row_invnorm<<<16, 256, 0, stream>>>(mu, invmu, NB * KK);
    k_zlogits<<<MROW / 8, 256, 0, stream>>>(mu, xe, invmu, invx, z);
    k_nz<<<16, 256, 0, stream>>>(z, nz);
    k_atx<<<NB * 8, 256, 0, stream>>>(nz, xe, mu);
  }

  // 3. random walk via Woodbury (exact)
  k_row_invnorm<<<16, 256, 0, stream>>>(mu, invmu, NB * KK);
  k_zlogits<<<MROW / 8, 256, 0, stream>>>(mu, xe, invmu, invx, z);
  k_nz<<<16, 256, 0, stream>>>(z, nz);
  k_smat<<<NB, 64, 0, stream>>>(nz, z, sinv);
  k_y2x<<<MROW, 256, 0, stream>>>(z, mu, xe, y2x);
  k_atx<<<NB * 8, 256, 0, stream>>>(nz, y2x, P);
  k_q<<<NB * 8, 256, 0, stream>>>(sinv, P, Q);
  k_refine<<<MROW, 256, 0, stream>>>(z, Q, y2x);  // y2x -> refined_x

  // 4. normalized centers
  k_centers<<<3, 256, 0, stream>>>(ac, fg, nacfg);

  // 5. prediction on original features
  k_frm<<<MROW / 8, 256, 0, stream>>>(xe, invx, nacfg, out + O_FRM, out + O_ATT, cwatt);
  k_att_sums<<<NB, 256, 0, stream>>>(out + O_ATT, cwatt, casum, cwsum);
  k_ca_feat<<<NB * 8, 256, 0, stream>>>(xe, out + O_ATT, casum, cafeat);
  k_cw_feat<<<NB * 8, 256, 0, stream>>>(xe, cwatt, cwsum, cwfeat);
  k_scores<<<NB, 256, 0, stream>>>(cafeat, cwfeat, nacfg, cascr, cwscr);
  k_softmax21<<<2, 256, 0, stream>>>(cascr, out + O_CA, NB);
  k_softmax21<<<2, 256, 0, stream>>>(cwscr, out + O_CW, NB);

  // 6. prediction on reallocated features
  k_row_invnorm<<<MROW / 8, 256, 0, stream>>>(y2x, invr, MROW);
  k_frm<<<MROW / 8, 256, 0, stream>>>(y2x, invr, nacfg, out + MO_FRM, out + MO_ATT, cwatt);
  k_att_sums<<<NB, 256, 0, stream>>>(out + MO_ATT, cwatt, casum, cwsum);
  k_ca_feat<<<NB * 8, 256, 0, stream>>>(y2x, out + MO_ATT, casum, cafeat);
  k_cw_feat<<<NB * 8, 256, 0, stream>>>(y2x, cwatt, cwsum, cwfeat);
  k_scores<<<NB, 256, 0, stream>>>(cafeat, cwfeat, nacfg, cascr, cwscr);
  k_softmax21<<<2, 256, 0, stream>>>(cascr, out + MO_CA, NB);
  k_softmax21<<<2, 256, 0, stream>>>(cwscr, out + MO_CW, NB);

  // 7. mu_pred
  k_mu_pred<<<16, 256, 0, stream>>>(mu, nacfg, out + MUP_OFF);
}
